// MultiHeadSelfAttention_69252052680851
// MI455X (gfx1250) — matrix operand order repaired and checked by reading, not compiled, not run on hardware
//
#include <hip/hip_runtime.h>
#include <hip/hip_bf16.h>

// ---------------------------------------------------------------------------
// Causal MHSA + RoPE for gfx1250 (MI455X), bf16 WMMA pipeline.
//   S=4096 seq, DM=1024 model, NH=16 heads, DK=64 head dim.
// Compute-bound (~103 GFLOP vs ~48MB traffic) -> all GEMMs run on
// v_wmma_f32_16x16x32_bf16 with fp32 accumulation. Projection GEMMs are
// M-blocked x4 so each weight B-fragment feeds 4 WMMAs (loads/WMMA ~0.58).
// ---------------------------------------------------------------------------

constexpr int S  = 4096;
constexpr int DM = 1024;
constexpr int NH = 16;
constexpr int DK = 64;

typedef __attribute__((ext_vector_type(16))) __bf16 v16bf;
typedef __attribute__((ext_vector_type(8)))  __bf16 v8bf;
typedef __attribute__((ext_vector_type(8)))  float  v8f;

// ---- WMMA wrapper ----------------------------------------------------------
__device__ __forceinline__ v8f wmma_bf16(v16bf a, v16bf b, v8f c) {
  return __builtin_amdgcn_wmma_f32_16x16x32_bf16(
      /*neg_a=*/false, a, /*neg_b=*/false, b,
      /*c_mod=*/(short)0, c, /*reuse_a=*/false, /*reuse_b=*/false);
}

// ---- Fragment loaders (layouts per CDNA5 ISA 7.12.2) -----------
// A 16x32 bf16: lane holds row M=lane%16; per lane the 16 elements cover
//   k = k0 + (hi?8:0) + {0..7} and k0 + (hi?8:0) + {16..23}  -> two 16B loads.
__device__ __forceinline__ v16bf ldA(const __bf16* base, int ld, int m0, int k0,
                                     int ln, int hi) {
  const __bf16* row = base + (size_t)(m0 + ln) * ld + k0 + (hi ? 8 : 0);
  v8bf a = *(const v8bf*)row;
  v8bf b = *(const v8bf*)(row + 16);
  v16bf r;
#pragma unroll
  for (int i = 0; i < 8; ++i) { r[i] = a[i]; r[i + 8] = b[i]; }
  return r;
}

// B 32x16 bf16 with B[k][n] = W[n][k] (row-major W, row stride ld):
// lane holds column N=lane%16; elements cover k = k0 + (hi?8:0) + {0..7}
// and k0 + (hi?8:0) + {16..23} -> two 16B loads.
__device__ __forceinline__ v16bf ldBT(const __bf16* base, int ld, int n0, int k0,
                                      int ln, int hi) {
  const __bf16* row = base + (size_t)(n0 + ln) * ld + k0 + (hi ? 8 : 0);
  v8bf a = *(const v8bf*)row;
  v8bf b = *(const v8bf*)(row + 16);
  v16bf r;
#pragma unroll
  for (int i = 0; i < 8; ++i) { r[i] = a[i]; r[i + 8] = b[i]; }
  return r;
}

// ---- fp32 -> bf16 conversion ----------------------------------------------
__global__ void cvt_bf16(const float* __restrict__ s, __bf16* __restrict__ d, int n) {
  int i = blockIdx.x * blockDim.x + threadIdx.x;
  if (i < n) d[i] = (__bf16)s[i];
}

// ---- Fused Q/K/V projection + RoPE ----------------------------------------
// One wave computes a 64x16 strip (4 row-tiles) for Q, K and V at once:
// per K-step the 3 weight B-fragments are reused by 4 A-fragments -> 12 WMMAs.
// Q,K (rotated) stored [h][s][64]; V stored transposed [h][64][s] so the
// P*V GEMM gets contiguous per-lane B fragments.
__global__ void __launch_bounds__(128) proj_qkv(
    const __bf16* __restrict__ xb,
    const __bf16* __restrict__ wq, const __bf16* __restrict__ wk,
    const __bf16* __restrict__ wv,
    __bf16* __restrict__ qo, __bf16* __restrict__ ko, __bf16* __restrict__ vto) {
  int wave = blockIdx.x * 4 + (threadIdx.x >> 5);
  int nt  = wave & 63;         // 64 column tiles (1024/16)
  int mtb = wave >> 6;         // 64 row blocks (4096/64)
  int m0 = mtb * 64, n0 = nt * 16;
  int lane = threadIdx.x & 31, ln = lane & 15, hi = lane >> 4;

  v8f aq[4] = {}, ak[4] = {}, av[4] = {};
  for (int k0 = 0; k0 < DM; k0 += 32) {
    if (k0 + 32 < DM) {
      __builtin_prefetch(xb + (size_t)(m0 + ln) * DM + k0 + 32, 0, 3);
      __builtin_prefetch(wq + (size_t)(n0 + ln) * DM + k0 + 32, 0, 3);
    }
    v16bf bq = ldBT(wq, DM, n0, k0, ln, hi);
    v16bf bk = ldBT(wk, DM, n0, k0, ln, hi);
    v16bf bv = ldBT(wv, DM, n0, k0, ln, hi);
#pragma unroll
    for (int mi = 0; mi < 4; ++mi) {
      v16bf af = ldA(xb, DM, m0 + 16 * mi, k0, ln, hi);
      aq[mi] = wmma_bf16(af, bq, aq[mi]);
      ak[mi] = wmma_bf16(af, bk, ak[mi]);
      av[mi] = wmma_bf16(af, bv, av[mi]);
    }
  }

  // RoPE: column n (head-dim d = n%64) pairs with n^1, which lives in the
  // adjacent lane of the same half -> shfl_xor mask 1.
  int head = n0 / DK;
  int dcol = (n0 % DK) + ln;               // 0..63 within head
  int i2   = dcol >> 1;                    // pair index
  float inv_freq = __powf(10000.0f, -(float)(2 * i2) / (float)DK);
  bool even = (dcol & 1) == 0;

#pragma unroll
  for (int mi = 0; mi < 4; ++mi) {
#pragma unroll
    for (int r = 0; r < 8; ++r) {
      int m = m0 + 16 * mi + r + 8 * hi;   // sequence position
      float sn, cs;
      __sincosf((float)m * inv_freq, &sn, &cs);
      float xq = aq[mi][r], pq = __shfl_xor(xq, 1, 32);
      float qr = even ? (xq * cs - pq * sn) : (pq * sn + xq * cs);
      float xk = ak[mi][r], pk = __shfl_xor(xk, 1, 32);
      float kr = even ? (xk * cs - pk * sn) : (pk * sn + xk * cs);

      size_t base = (size_t)head * S * DK + (size_t)m * DK + (n0 % DK) + ln;
      qo[base] = (__bf16)qr;
      ko[base] = (__bf16)kr;
      vto[(size_t)head * DK * S + (size_t)((n0 % DK) + ln) * S + m] =
          (__bf16)av[mi][r];
    }
  }
}

// ---- Flash attention (causal, online softmax) -----------------------------
// One wave per (head, 16-query tile); 32-key chunks:
//   4 WMMAs for S = Q*K^T (two 16-col tiles x K=64), softmax update,
//   P staged through LDS (C/D -> A layout), 4 WMMAs for ctx += P*V.
__global__ void __launch_bounds__(128) attn(
    const __bf16* __restrict__ q, const __bf16* __restrict__ k,
    const __bf16* __restrict__ vt, __bf16* __restrict__ ctx) {
  __shared__ __align__(16) __bf16 lds_p[4][16 * 32];

  int wslot = threadIdx.x >> 5;
  int wave  = blockIdx.x * 4 + wslot;
  int head  = wave >> 8;                   // 256 q-tiles per head
  int q0    = (wave & 255) * 16;
  int lane = threadIdx.x & 31, ln = lane & 15, hi = lane >> 4;

  const __bf16* qh = q  + (size_t)head * S * DK;
  const __bf16* kh = k  + (size_t)head * S * DK;
  const __bf16* vh = vt + (size_t)head * DK * S;
  __bf16* lp = lds_p[wslot];

  v16bf a0 = ldA(qh, DK, q0, 0,  ln, hi);  // Q tile, d = 0..31
  v16bf a1 = ldA(qh, DK, q0, 32, ln, hi);  // Q tile, d = 32..63

  v8f acc0 = {}, acc1 = {}, acc2 = {}, acc3 = {};
  float mrow[8], lrow[8];
#pragma unroll
  for (int r = 0; r < 8; ++r) { mrow[r] = -1e30f; lrow[r] = 0.0f; }

  const float scale = 0.125f;              // 1/sqrt(64)
  int qend = q0 + 15;
  for (int j0 = 0; j0 <= qend; j0 += 32) {
    if (j0 + 32 <= qend)
      __builtin_prefetch(kh + (size_t)(j0 + 32 + ln) * DK, 0, 3);

    v8f s0 = {}, s1 = {};
    s0 = wmma_bf16(a0, ldBT(kh, DK, j0,      0,  ln, hi), s0);
    s0 = wmma_bf16(a1, ldBT(kh, DK, j0,      32, ln, hi), s0);
    s1 = wmma_bf16(a0, ldBT(kh, DK, j0 + 16, 0,  ln, hi), s1);
    s1 = wmma_bf16(a1, ldBT(kh, DK, j0 + 16, 32, ln, hi), s1);

#pragma unroll
    for (int r = 0; r < 8; ++r) {
      int m = q0 + r + 8 * hi;
      // scale + branch-free causal mask (keeps EXEC all-ones for WMMA)
      float v0 = (j0 + ln      > m) ? -1e30f : s0[r] * scale;
      float v1 = (j0 + 16 + ln > m) ? -1e30f : s1[r] * scale;

      float mx = fmaxf(v0, v1);
      mx = fmaxf(mx, __shfl_xor(mx, 1, 32));
      mx = fmaxf(mx, __shfl_xor(mx, 2, 32));
      mx = fmaxf(mx, __shfl_xor(mx, 4, 32));
      mx = fmaxf(mx, __shfl_xor(mx, 8, 32));
      float newm = fmaxf(mrow[r], mx);
      float corr = __expf(mrow[r] - newm);
      float p0 = __expf(v0 - newm);
      float p1 = __expf(v1 - newm);
      float rs = p0 + p1;
      rs += __shfl_xor(rs, 1, 32);
      rs += __shfl_xor(rs, 2, 32);
      rs += __shfl_xor(rs, 4, 32);
      rs += __shfl_xor(rs, 8, 32);
      lrow[r] = lrow[r] * corr + rs;
      mrow[r] = newm;
      acc0[r] *= corr; acc1[r] *= corr; acc2[r] *= corr; acc3[r] *= corr;

      int row = r + 8 * hi;                // C/D layout -> LDS row-major P
      lp[row * 32 + ln]      = (__bf16)p0;
      lp[row * 32 + 16 + ln] = (__bf16)p1;
    }

    // ctx(16x64) += P(16x32) * V(32x64); Vt rows are head-dims, j contiguous.
    v16bf ap = ldA(lp, 32, 0, 0, ln, hi);  // per-wave LDS, in-order DS ops
    acc0 = wmma_bf16(ap, ldBT(vh, S, 0,  j0, ln, hi), acc0);
    acc1 = wmma_bf16(ap, ldBT(vh, S, 16, j0, ln, hi), acc1);
    acc2 = wmma_bf16(ap, ldBT(vh, S, 32, j0, ln, hi), acc2);
    acc3 = wmma_bf16(ap, ldBT(vh, S, 48, j0, ln, hi), acc3);
  }

  // Normalize and store ctx as bf16 [s][1024] for the output projection.
#pragma unroll
  for (int r = 0; r < 8; ++r) {
    float inv = 1.0f / lrow[r];
    int m = q0 + r + 8 * hi;
    size_t base = (size_t)m * DM + head * DK;
    ctx[base + 0  + ln] = (__bf16)(acc0[r] * inv);
    ctx[base + 16 + ln] = (__bf16)(acc1[r] * inv);
    ctx[base + 32 + ln] = (__bf16)(acc2[r] * inv);
    ctx[base + 48 + ln] = (__bf16)(acc3[r] * inv);
  }
}

// ---- Output projection: out = ctx * Wo^T (fp32 out), M-blocked x4 ---------
__global__ void __launch_bounds__(128) proj_out(
    const __bf16* __restrict__ ctx, const __bf16* __restrict__ wo,
    float* __restrict__ out) {
  int wave = blockIdx.x * 4 + (threadIdx.x >> 5);
  int nt = wave & 63, mtb = wave >> 6;
  int m0 = mtb * 64, n0 = nt * 16;
  int lane = threadIdx.x & 31, ln = lane & 15, hi = lane >> 4;

  v8f acc[4] = {};
  for (int k0 = 0; k0 < DM; k0 += 32) {
    if (k0 + 32 < DM) {
      __builtin_prefetch(ctx + (size_t)(m0 + ln) * DM + k0 + 32, 0, 3);
      __builtin_prefetch(wo + (size_t)(n0 + ln) * DM + k0 + 32, 0, 3);
    }
    v16bf b = ldBT(wo, DM, n0, k0, ln, hi);
#pragma unroll
    for (int mi = 0; mi < 4; ++mi) {
      v16bf a = ldA(ctx, DM, m0 + 16 * mi, k0, ln, hi);
      acc[mi] = wmma_bf16(a, b, acc[mi]);
    }
  }
#pragma unroll
  for (int mi = 0; mi < 4; ++mi)
#pragma unroll
    for (int r = 0; r < 8; ++r)
      out[(size_t)(m0 + 16 * mi + r + 8 * hi) * DM + n0 + ln] = acc[mi][r];
}

// ---------------------------------------------------------------------------
extern "C" void kernel_launch(void* const* d_in, const int* in_sizes, int n_in,
                              void* d_out, int out_size, void* d_ws, size_t ws_size,
                              hipStream_t stream) {
  const float* x  = (const float*)d_in[0];
  const float* Wq = (const float*)d_in[1];
  const float* Wk = (const float*)d_in[2];
  const float* Wv = (const float*)d_in[3];
  const float* Wo = (const float*)d_in[4];
  float* out = (float*)d_out;

  // Workspace layout (bf16 elements): 48 MiB total.
  __bf16* ws   = (__bf16*)d_ws;
  __bf16* xb   = ws;                                  // 4096*1024
  __bf16* wqb  = xb  + (size_t)S * DM;                // 1024*1024
  __bf16* wkb  = wqb + (size_t)DM * DM;
  __bf16* wvb  = wkb + (size_t)DM * DM;
  __bf16* wob  = wvb + (size_t)DM * DM;
  __bf16* qb   = wob + (size_t)DM * DM;               // [16][4096][64]
  __bf16* kb   = qb  + (size_t)S * DM;
  __bf16* vtb  = kb  + (size_t)S * DM;                // [16][64][4096]
  __bf16* ctxb = vtb + (size_t)S * DM;                // [4096][1024]

  const int nX = S * DM, nW = DM * DM;
  cvt_bf16<<<nX / 256, 256, 0, stream>>>(x,  xb,  nX);
  cvt_bf16<<<nW / 256, 256, 0, stream>>>(Wq, wqb, nW);
  cvt_bf16<<<nW / 256, 256, 0, stream>>>(Wk, wkb, nW);
  cvt_bf16<<<nW / 256, 256, 0, stream>>>(Wv, wvb, nW);
  cvt_bf16<<<nW / 256, 256, 0, stream>>>(Wo, wob, nW);

  // (S/64)*(DM/16) = 4096 wave-strips, 4 waves/block.
  proj_qkv<<<(64 * 64) / 4, 128, 0, stream>>>(xb, wqb, wkb, wvb, qb, kb, vtb);

  // NH * (S/16) = 4096 wave-units, 4 waves/block.
  attn<<<(NH * 256) / 4, 128, 0, stream>>>(qb, kb, vtb, ctxb);

  proj_out<<<(64 * 64) / 4, 128, 0, stream>>>(ctxb, wob, out);
}
